// ZetaMixup_30777735643560
// MI455X (gfx1250) — matrix operand order repaired and checked by reading, not compiled, not run on hardware
//
#include <hip/hip_runtime.h>
#include <hip/hip_bf16.h>

#define S_SAMPLES 1024
#define BDIM      512        // B == L == K of the GEMM
#define DFEAT     12288      // 3*64*64
#define NCLS      100
#define GAMMA_F   2.4f

typedef __attribute__((ext_vector_type(16))) _Float16 v16h;
typedef __attribute__((ext_vector_type(8)))  _Float16 v8h;
typedef __attribute__((ext_vector_type(4)))  _Float16 v4h;
typedef __attribute__((ext_vector_type(8)))  float    v8f;

// Keyed bijection on [0,512): 10-bit balanced Feistel, cycle-walked.
// Bijective per sample key => Wmat rows are exact permutations of w[].
__device__ __forceinline__ unsigned perm512(unsigned s, unsigned x) {
  do {
    unsigned l = x & 31u, r = (x >> 5) & 31u;
#pragma unroll
    for (unsigned i = 0; i < 4u; ++i) {
      unsigned f = (r * 0x9E3779B1u) ^ ((s + 0x85EBCA77u) * (0xC2B2AE35u + i * 0x27D4EB2Fu));
      f = (f >> 13) & 31u;
      unsigned t = l ^ f;
      l = r; r = t;
    }
    x = (r << 5) | l;
  } while (x >= 512u);
  return x;
}

// ---- setup: w[l] = (l+1)^-gamma, invCn = 1/sum(w) -------------------------
__global__ void zeta_init(float* __restrict__ wsf) {
  __shared__ float red[512];
  const int l = threadIdx.x;
  float wv = __powf((float)(l + 1), -GAMMA_F);
  wsf[l] = wv;
  red[l] = wv;
  __syncthreads();
  for (int off = 256; off > 0; off >>= 1) {
    if (l < off) red[l] += red[l + off];
    __syncthreads();
  }
  if (l == 0) wsf[512] = 1.0f / red[0];
}

// ---- build Wmat in f16 (scatter via bijection: every slot written once) ---
__global__ void zeta_wmat(const float* __restrict__ wsf, _Float16* __restrict__ Wh) {
  const unsigned idx = blockIdx.x * blockDim.x + threadIdx.x;  // s*512 + l
  const unsigned s = idx >> 9, l = idx & 511u;
  const unsigned p = perm512(s, l);
  Wh[((size_t)s << 9) + p] = (_Float16)wsf[l];
}

// ---- cast X to f16 --------------------------------------------------------
__global__ void zeta_cast(const float* __restrict__ X, _Float16* __restrict__ Xh) {
  const size_t i = ((size_t)blockIdx.x * blockDim.x + threadIdx.x) * 4u;
  float4 v = *(const float4*)(X + i);
  v4h o = { (_Float16)v.x, (_Float16)v.y, (_Float16)v.z, (_Float16)v.w };
  *(v4h*)(Xh + i) = o;
}

// ---- mix_x = (Wmat @ X) * invCn via v_wmma_f32_16x16x32_f16 ---------------
// grid = (S/16, D/512), block = 256 (8 waves). Wave w computes the 16x64
// tile at (m0, n0 + 64*w); 4 accumulators, 4 WMMAs per K-step, 64 total.
__global__ __launch_bounds__(256)
void zeta_gemm(const _Float16* __restrict__ Wh, const _Float16* __restrict__ Xh,
               const float* __restrict__ wsf, float* __restrict__ outX) {
  const float invCn = wsf[512];
  const unsigned lane = threadIdx.x & 31u;
  const unsigned wave = threadIdx.x >> 5;
  const unsigned m0 = blockIdx.x * 16u;
  const unsigned n0 = blockIdx.y * 512u + wave * 64u;

  v8f c0 = {}, c1 = {}, c2 = {}, c3 = {};

  // A fragment addressing (16x32 f16 layout, ISA 7.12.2)
  const unsigned arow = m0 + (lane & 15u);
  const unsigned koff = (lane >> 4) * 8u;
  const _Float16* aptr = Wh + (size_t)arow * BDIM + koff;
  // B fragment addressing: lane l -> N = l&15, elements over K in A's order
  const _Float16* bptr = Xh + n0 + (lane & 15u);

  for (unsigned k0 = 0; k0 < BDIM; k0 += 32u) {
    v8h alo = *(const v8h*)(aptr);
    v8h ahi = *(const v8h*)(aptr + 16);
    v16h a;
#pragma unroll
    for (int i = 0; i < 8; ++i) { a[i] = alo[i]; a[8 + i] = ahi[i]; }

    v16h b0, b1, b2, b3;
#pragma unroll
    for (int i = 0; i < 16; ++i) {
      const unsigned kk = (i < 8) ? (koff + i) : (16u + koff + (i - 8));
      const _Float16* bp = bptr + (size_t)kk * DFEAT;
      b0[i] = bp[0]; b1[i] = bp[16]; b2[i] = bp[32]; b3[i] = bp[48];
    }

    c0 = __builtin_amdgcn_wmma_f32_16x16x32_f16(false, a, false, b0, (short)0, c0, false, false);
    c1 = __builtin_amdgcn_wmma_f32_16x16x32_f16(false, a, false, b1, (short)0, c1, false, false);
    c2 = __builtin_amdgcn_wmma_f32_16x16x32_f16(false, a, false, b2, (short)0, c2, false, false);
    c3 = __builtin_amdgcn_wmma_f32_16x16x32_f16(false, a, false, b3, (short)0, c3, false, false);

    aptr += 32u;
    bptr += (size_t)32u * DFEAT;
  }

  // C/D layout: lane l, vgpr v -> M = v + 8*(l>>4), N = l&15
  const unsigned orow = m0 + 8u * (lane >> 4);
  const unsigned ocol = n0 + (lane & 15u);
  float* out = outX + (size_t)orow * DFEAT + ocol;
#pragma unroll
  for (int v = 0; v < 8; ++v) {
    out[(size_t)v * DFEAT +  0] = c0[v] * invCn;
    out[(size_t)v * DFEAT + 16] = c1[v] * invCn;
    out[(size_t)v * DFEAT + 32] = c2[v] * invCn;
    out[(size_t)v * DFEAT + 48] = c3[v] * invCn;
  }
}

// ---- mix_y[s,c] = invCn * sum_l w[l] * logits[l, p_s(l), c] ---------------
// One block per sample; stage w[] and the permutation in LDS; threads over c
// give coalesced 100-float rows per (l, p) gather.
__global__ __launch_bounds__(128)
void zeta_mixy(const float* __restrict__ logits, const float* __restrict__ wsf,
               float* __restrict__ outY) {
  __shared__ float sw[512];
  __shared__ int   sp[512];
  const unsigned s = blockIdx.x;
  for (unsigned l = threadIdx.x; l < 512u; l += 128u) {
    sw[l] = wsf[l];
    sp[l] = (int)perm512(s, l);
  }
  __syncthreads();
  const float invCn = wsf[512];
  const unsigned c = threadIdx.x;
  if (c < NCLS) {
    float acc = 0.f;
#pragma unroll 4
    for (int l = 0; l < 512; ++l)
      acc += sw[l] * logits[((size_t)l * 512u + (unsigned)sp[l]) * NCLS + c];
    outY[(size_t)s * NCLS + c] = acc * invCn;
  }
}

extern "C" void kernel_launch(void* const* d_in, const int* in_sizes, int n_in,
                              void* d_out, int out_size, void* d_ws, size_t ws_size,
                              hipStream_t stream) {
  const float* X      = (const float*)d_in[0];
  const float* logits = (const float*)d_in[1];
  (void)in_sizes; (void)n_in; (void)out_size; (void)ws_size;

  // workspace layout: [0..511] w, [512] invCn, pad to 4KB,
  // then Wmat f16 (1 MB), then X f16 (12.6 MB)
  float*    wsf = (float*)d_ws;
  _Float16* Wh  = (_Float16*)((char*)d_ws + 4096);
  _Float16* Xh  = (_Float16*)((char*)d_ws + 4096 + (size_t)S_SAMPLES * BDIM * 2);

  float* outX = (float*)d_out;
  float* outY = outX + (size_t)S_SAMPLES * DFEAT;

  zeta_init<<<1, 512, 0, stream>>>(wsf);
  zeta_wmat<<<(S_SAMPLES * BDIM) / 256, 256, 0, stream>>>(wsf, Wh);
  zeta_cast<<<((size_t)BDIM * DFEAT) / (4 * 256), 256, 0, stream>>>(X, Xh);
  dim3 g(S_SAMPLES / 16, DFEAT / 512);
  zeta_gemm<<<g, 256, 0, stream>>>(Wh, Xh, wsf, outX);
  zeta_mixy<<<S_SAMPLES, 128, 0, stream>>>(logits, wsf, outY);
}
